// HybridLinearizationRnn_17231408791960
// MI455X (gfx1250) — matrix operand order repaired and checked by reading, not compiled, not run on hardware
//
#include <hip/hip_runtime.h>
#include <hip/hip_bf16.h>

// Problem sizes (fixed by the reference).
#define B_SZ 512
#define N_T  512
#define NX   512
#define NU   256
#define NY   256
#define NW   256
#define BM   16      // batch rows per workgroup == WMMA M
#define NTHREADS 256 // 8 wave32s

typedef __attribute__((ext_vector_type(16))) __bf16 v16bf;
typedef __attribute__((ext_vector_type(8)))  __bf16 v8bf;
typedef __attribute__((ext_vector_type(4)))  __bf16 v4bf;
typedef __attribute__((ext_vector_type(8)))  float  v8f;

// ---------------------------------------------------------------------------
// A-fragment: 16x32 bf16 from a row-major LDS tile (rows = batch rows).
// 16-bit A layout: lane m (0-15) holds row m; halves map to
//   K = g*16 + hi*8 + j   (g = half-group, hi = lane>>4, j = 0..7)
// ---------------------------------------------------------------------------
__device__ inline v16bf load_a_frag(const __bf16* base, int stride, int kb, int lane) {
    const int m  = lane & 15;
    const int hi = lane >> 4;
    union { v16bf v; v8bf h[2]; } u;
    const __bf16* p = base + m * stride + kb + hi * 8;
    u.h[0] = *(const v8bf*)(p);
    u.h[1] = *(const v8bf*)(p + 16);
    return u.v;
}

// ---------------------------------------------------------------------------
// B-fragment: 32x16 (KxN). Weights kept O x K row-major, so B column n is
// weight row (oBase+n): K = g*16 + hi*8 + j (same as A fragment).
// ---------------------------------------------------------------------------
__device__ inline v16bf load_b_frag(const __bf16* w, int K, int oBase, int kb, int lane) {
    const int n  = lane & 15;
    const int hi = lane >> 4;
    union { v16bf v; v8bf h[2]; } u;
    const __bf16* p = w + (size_t)(oBase + n) * K + kb + hi * 8;
    u.h[0] = *(const v8bf*)(p);
    u.h[1] = *(const v8bf*)(p + 16);
    return u.v;
}

__device__ inline v8f wmma_bf16(v16bf a, v16bf b, v8f c) {
    return __builtin_amdgcn_wmma_f32_16x16x32_bf16(
        false, a, false, b, (short)0, c, false, false);
}

__device__ inline v4bf cvt4(float4 f) {
    v4bf r;
    r[0] = (__bf16)f.x; r[1] = (__bf16)f.y; r[2] = (__bf16)f.z; r[3] = (__bf16)f.w;
    return r;
}

// f32 -> bf16 weight conversion (prep pass; O x K row-major preserved).
__global__ void f32_to_bf16_kernel(const float* __restrict__ src,
                                   __bf16* __restrict__ dst, int n) {
    int i = blockIdx.x * blockDim.x + threadIdx.x;
    if (i < n) dst[i] = (__bf16)src[i];
}

// ===========================================================================
// Split path, kernel 1: serial recurrence only (w, x'); spills x_k, w_k bf16
// with coalesced uint4 copies overlapped with u staging. 32 persistent
// workgroups (one per 16-row batch tile), 8 waves each.
// ===========================================================================
__global__ __launch_bounds__(NTHREADS)
void lure_rec_kernel(const float* __restrict__ x0,
                     const float* __restrict__ us,
                     const __bf16* __restrict__ Aw,
                     const __bf16* __restrict__ B1w,
                     const __bf16* __restrict__ B2w,
                     const __bf16* __restrict__ C2w,
                     const __bf16* __restrict__ D21w,
                     __bf16* __restrict__ Xst,   // (N, B, NX) bf16
                     __bf16* __restrict__ Wst) { // (N, B, NW) bf16
    __shared__ __bf16 sX[2][BM][NX];
    __shared__ __bf16 sU[BM][NU];
    __shared__ __bf16 sW[BM][NW];

    const int tid   = threadIdx.x;
    const int wave  = tid >> 5;
    const int lane  = tid & 31;
    const int bBase = blockIdx.x * BM;
    const int n     = lane & 15;
    const int hi    = lane >> 4;

    // Stage x0 -> sX[0] (bf16).
    for (int i = tid; i < BM * NX / 4; i += NTHREADS) {
        const int r = i / (NX / 4), c4 = i % (NX / 4);
        float4 f = ((const float4*)(x0 + (size_t)(bBase + r) * NX))[c4];
        *(v4bf*)&sX[0][r][c4 * 4] = cvt4(f);
    }
    __syncthreads();

    int cur = 0;
    for (int k = 0; k < N_T; ++k) {
        // Stage u_k (f32 -> bf16, vectorized); prefetch next step's rows.
        for (int i = tid; i < BM * NU / 4; i += NTHREADS) {
            const int r = i / (NU / 4), c4 = i % (NU / 4);
            float4 f = ((const float4*)(us + ((size_t)(bBase + r) * N_T + k) * NU))[c4];
            *(v4bf*)&sU[r][c4 * 4] = cvt4(f);
        }
        // Coalesced spill of x_k (sitting in sX[cur]) and w_{k-1} (in sW).
        {
            const uint4* sxv = (const uint4*)&sX[cur][0][0];
            uint4* xdst = (uint4*)(Xst + ((size_t)k * B_SZ + bBase) * NX);
            for (int i = tid; i < BM * NX / 8; i += NTHREADS) xdst[i] = sxv[i];
            if (k > 0) {
                const uint4* swv = (const uint4*)&sW[0][0];
                uint4* wdst = (uint4*)(Wst + ((size_t)(k - 1) * B_SZ + bBase) * NW);
                for (int i = tid; i < BM * NW / 8; i += NTHREADS) wdst[i] = swv[i];
            }
        }
        if (k + 1 < N_T && tid < BM)
            __builtin_prefetch(&us[((size_t)(bBase + tid) * N_T + (k + 1)) * NU], 0, 1);
        __syncthreads();

        // ---- w = tanh(x@C2^T + u@D21^T): one tile-pair per wave -----------
        {
            const int o0 = wave * 32, o1 = o0 + 16;
            v8f a0 = {}, a1 = {};
            #pragma unroll 4
            for (int kb = 0; kb < NX; kb += 32) {
                v16bf a = load_a_frag(&sX[cur][0][0], NX, kb, lane);
                a0 = wmma_bf16(a, load_b_frag(C2w, NX, o0, kb, lane), a0);
                a1 = wmma_bf16(a, load_b_frag(C2w, NX, o1, kb, lane), a1);
            }
            #pragma unroll 4
            for (int kb = 0; kb < NU; kb += 32) {
                v16bf a = load_a_frag(&sU[0][0], NU, kb, lane);
                a0 = wmma_bf16(a, load_b_frag(D21w, NU, o0, kb, lane), a0);
                a1 = wmma_bf16(a, load_b_frag(D21w, NU, o1, kb, lane), a1);
            }
            #pragma unroll
            for (int r = 0; r < 8; ++r) {
                const int m = hi * 8 + r;
                sW[m][o0 + n] = (__bf16)tanhf(a0[r]);
                sW[m][o1 + n] = (__bf16)tanhf(a1[r]);
            }
        }
        __syncthreads();

        // ---- x' = x@A^T + u@B1^T + w@B2^T: two tile-pairs per wave --------
        const int nxt = cur ^ 1;
        for (int p = 0; p < 2; ++p) {
            const int o0 = wave * 64 + p * 32, o1 = o0 + 16;
            v8f a0 = {}, a1 = {};
            #pragma unroll 4
            for (int kb = 0; kb < NX; kb += 32) {
                v16bf a = load_a_frag(&sX[cur][0][0], NX, kb, lane);
                a0 = wmma_bf16(a, load_b_frag(Aw, NX, o0, kb, lane), a0);
                a1 = wmma_bf16(a, load_b_frag(Aw, NX, o1, kb, lane), a1);
            }
            #pragma unroll 4
            for (int kb = 0; kb < NU; kb += 32) {
                v16bf a = load_a_frag(&sU[0][0], NU, kb, lane);
                a0 = wmma_bf16(a, load_b_frag(B1w, NU, o0, kb, lane), a0);
                a1 = wmma_bf16(a, load_b_frag(B1w, NU, o1, kb, lane), a1);
            }
            #pragma unroll 4
            for (int kb = 0; kb < NW; kb += 32) {
                v16bf a = load_a_frag(&sW[0][0], NW, kb, lane);
                a0 = wmma_bf16(a, load_b_frag(B2w, NW, o0, kb, lane), a0);
                a1 = wmma_bf16(a, load_b_frag(B2w, NW, o1, kb, lane), a1);
            }
            #pragma unroll
            for (int r = 0; r < 8; ++r) {
                const int m = hi * 8 + r;
                sX[nxt][m][o0 + n] = (__bf16)a0[r];
                sX[nxt][m][o1 + n] = (__bf16)a1[r];
            }
        }
        __syncthreads();
        cur = nxt;
    }

    // Spill the last w (w_{N-1}).
    {
        const uint4* swv = (const uint4*)&sW[0][0];
        uint4* wdst = (uint4*)(Wst + ((size_t)(N_T - 1) * B_SZ + bBase) * NW);
        for (int i = tid; i < BM * NW / 8; i += NTHREADS) wdst[i] = swv[i];
    }
}

// ===========================================================================
// Split path, kernel 2: y = x@C1^T + u@D11^T + w@D12^T, fully parallel over
// (k, batch-tile): 512 x 32 = 16384 workgroups -> saturates the chip.
// ===========================================================================
__global__ __launch_bounds__(NTHREADS)
void lure_y_kernel(const float* __restrict__ us,
                   const __bf16* __restrict__ Xst,
                   const __bf16* __restrict__ Wst,
                   const __bf16* __restrict__ C1w,
                   const __bf16* __restrict__ D11w,
                   const __bf16* __restrict__ D12w,
                   float* __restrict__ ys) {
    __shared__ __bf16 sX[BM][NX];
    __shared__ __bf16 sU[BM][NU];
    __shared__ __bf16 sW[BM][NW];

    const int tid   = threadIdx.x;
    const int wave  = tid >> 5;
    const int lane  = tid & 31;
    const int bBase = blockIdx.x * BM;
    const int k     = blockIdx.y;
    const int n     = lane & 15;
    const int hi    = lane >> 4;

    // Stage x_k, w_k (bf16, contiguous) and u_k (f32 -> bf16).
    {
        const uint4* src = (const uint4*)(Xst + ((size_t)k * B_SZ + bBase) * NX);
        uint4* dst = (uint4*)&sX[0][0];
        for (int i = tid; i < BM * NX / 8; i += NTHREADS) dst[i] = src[i];
        const uint4* srcw = (const uint4*)(Wst + ((size_t)k * B_SZ + bBase) * NW);
        uint4* dstw = (uint4*)&sW[0][0];
        for (int i = tid; i < BM * NW / 8; i += NTHREADS) dstw[i] = srcw[i];
        for (int i = tid; i < BM * NU / 4; i += NTHREADS) {
            const int r = i / (NU / 4), c4 = i % (NU / 4);
            float4 f = ((const float4*)(us + ((size_t)(bBase + r) * N_T + k) * NU))[c4];
            *(v4bf*)&sU[r][c4 * 4] = cvt4(f);
        }
    }
    __syncthreads();

    const int o0 = wave * 32, o1 = o0 + 16;
    v8f a0 = {}, a1 = {};
    #pragma unroll 4
    for (int kb = 0; kb < NX; kb += 32) {
        v16bf a = load_a_frag(&sX[0][0], NX, kb, lane);
        a0 = wmma_bf16(a, load_b_frag(C1w, NX, o0, kb, lane), a0);
        a1 = wmma_bf16(a, load_b_frag(C1w, NX, o1, kb, lane), a1);
    }
    #pragma unroll 4
    for (int kb = 0; kb < NU; kb += 32) {
        v16bf a = load_a_frag(&sU[0][0], NU, kb, lane);
        a0 = wmma_bf16(a, load_b_frag(D11w, NU, o0, kb, lane), a0);
        a1 = wmma_bf16(a, load_b_frag(D11w, NU, o1, kb, lane), a1);
    }
    #pragma unroll 4
    for (int kb = 0; kb < NW; kb += 32) {
        v16bf a = load_a_frag(&sW[0][0], NW, kb, lane);
        a0 = wmma_bf16(a, load_b_frag(D12w, NW, o0, kb, lane), a0);
        a1 = wmma_bf16(a, load_b_frag(D12w, NW, o1, kb, lane), a1);
    }
    #pragma unroll
    for (int r = 0; r < 8; ++r) {
        const int m = hi * 8 + r;
        ys[((size_t)(bBase + m) * N_T + k) * NY + o0 + n] = a0[r];
        ys[((size_t)(bBase + m) * N_T + k) * NY + o1 + n] = a1[r];
    }
}

// ===========================================================================
// Fallback: fused persistent kernel (used when workspace can't hold spills).
// ===========================================================================
__global__ __launch_bounds__(NTHREADS)
void lure_fused_kernel(const float* __restrict__ x0,
                       const float* __restrict__ us,
                       float* __restrict__ ys,
                       const __bf16* __restrict__ Aw,
                       const __bf16* __restrict__ B1w,
                       const __bf16* __restrict__ B2w,
                       const __bf16* __restrict__ C1w,
                       const __bf16* __restrict__ D11w,
                       const __bf16* __restrict__ D12w,
                       const __bf16* __restrict__ C2w,
                       const __bf16* __restrict__ D21w) {
    __shared__ __bf16 sX[2][BM][NX];
    __shared__ __bf16 sU[BM][NU];
    __shared__ __bf16 sW[BM][NW];

    const int tid   = threadIdx.x;
    const int wave  = tid >> 5;
    const int lane  = tid & 31;
    const int bBase = blockIdx.x * BM;
    const int n     = lane & 15;
    const int hi    = lane >> 4;

    for (int i = tid; i < BM * NX / 4; i += NTHREADS) {
        const int r = i / (NX / 4), c4 = i % (NX / 4);
        float4 f = ((const float4*)(x0 + (size_t)(bBase + r) * NX))[c4];
        *(v4bf*)&sX[0][r][c4 * 4] = cvt4(f);
    }
    __syncthreads();

    int cur = 0;
    for (int k = 0; k < N_T; ++k) {
        for (int i = tid; i < BM * NU / 4; i += NTHREADS) {
            const int r = i / (NU / 4), c4 = i % (NU / 4);
            float4 f = ((const float4*)(us + ((size_t)(bBase + r) * N_T + k) * NU))[c4];
            *(v4bf*)&sU[r][c4 * 4] = cvt4(f);
        }
        if (k + 1 < N_T && tid < BM)
            __builtin_prefetch(&us[((size_t)(bBase + tid) * N_T + (k + 1)) * NU], 0, 1);
        __syncthreads();

        {   // w tiles: one pair per wave
            const int o0 = wave * 32, o1 = o0 + 16;
            v8f a0 = {}, a1 = {};
            #pragma unroll 4
            for (int kb = 0; kb < NX; kb += 32) {
                v16bf a = load_a_frag(&sX[cur][0][0], NX, kb, lane);
                a0 = wmma_bf16(a, load_b_frag(C2w, NX, o0, kb, lane), a0);
                a1 = wmma_bf16(a, load_b_frag(C2w, NX, o1, kb, lane), a1);
            }
            #pragma unroll 4
            for (int kb = 0; kb < NU; kb += 32) {
                v16bf a = load_a_frag(&sU[0][0], NU, kb, lane);
                a0 = wmma_bf16(a, load_b_frag(D21w, NU, o0, kb, lane), a0);
                a1 = wmma_bf16(a, load_b_frag(D21w, NU, o1, kb, lane), a1);
            }
            #pragma unroll
            for (int r = 0; r < 8; ++r) {
                const int m = hi * 8 + r;
                sW[m][o0 + n] = (__bf16)tanhf(a0[r]);
                sW[m][o1 + n] = (__bf16)tanhf(a1[r]);
            }
        }
        __syncthreads();

        const int nxt = cur ^ 1;
        for (int p = 0; p < 2; ++p) {   // x' tiles: two pairs per wave
            const int o0 = wave * 64 + p * 32, o1 = o0 + 16;
            v8f a0 = {}, a1 = {};
            #pragma unroll 4
            for (int kb = 0; kb < NX; kb += 32) {
                v16bf a = load_a_frag(&sX[cur][0][0], NX, kb, lane);
                a0 = wmma_bf16(a, load_b_frag(Aw, NX, o0, kb, lane), a0);
                a1 = wmma_bf16(a, load_b_frag(Aw, NX, o1, kb, lane), a1);
            }
            #pragma unroll 4
            for (int kb = 0; kb < NU; kb += 32) {
                v16bf a = load_a_frag(&sU[0][0], NU, kb, lane);
                a0 = wmma_bf16(a, load_b_frag(B1w, NU, o0, kb, lane), a0);
                a1 = wmma_bf16(a, load_b_frag(B1w, NU, o1, kb, lane), a1);
            }
            #pragma unroll 4
            for (int kb = 0; kb < NW; kb += 32) {
                v16bf a = load_a_frag(&sW[0][0], NW, kb, lane);
                a0 = wmma_bf16(a, load_b_frag(B2w, NW, o0, kb, lane), a0);
                a1 = wmma_bf16(a, load_b_frag(B2w, NW, o1, kb, lane), a1);
            }
            #pragma unroll
            for (int r = 0; r < 8; ++r) {
                const int m = hi * 8 + r;
                sX[nxt][m][o0 + n] = (__bf16)a0[r];
                sX[nxt][m][o1 + n] = (__bf16)a1[r];
            }
        }

        {   // y tiles: one pair per wave
            const int o0 = wave * 32, o1 = o0 + 16;
            v8f a0 = {}, a1 = {};
            #pragma unroll 4
            for (int kb = 0; kb < NX; kb += 32) {
                v16bf a = load_a_frag(&sX[cur][0][0], NX, kb, lane);
                a0 = wmma_bf16(a, load_b_frag(C1w, NX, o0, kb, lane), a0);
                a1 = wmma_bf16(a, load_b_frag(C1w, NX, o1, kb, lane), a1);
            }
            #pragma unroll 4
            for (int kb = 0; kb < NU; kb += 32) {
                v16bf a = load_a_frag(&sU[0][0], NU, kb, lane);
                a0 = wmma_bf16(a, load_b_frag(D11w, NU, o0, kb, lane), a0);
                a1 = wmma_bf16(a, load_b_frag(D11w, NU, o1, kb, lane), a1);
            }
            #pragma unroll 4
            for (int kb = 0; kb < NW; kb += 32) {
                v16bf a = load_a_frag(&sW[0][0], NW, kb, lane);
                a0 = wmma_bf16(a, load_b_frag(D12w, NW, o0, kb, lane), a0);
                a1 = wmma_bf16(a, load_b_frag(D12w, NW, o1, kb, lane), a1);
            }
            #pragma unroll
            for (int r = 0; r < 8; ++r) {
                const int m = hi * 8 + r;
                ys[((size_t)(bBase + m) * N_T + k) * NY + o0 + n] = a0[r];
                ys[((size_t)(bBase + m) * N_T + k) * NY + o1 + n] = a1[r];
            }
        }
        __syncthreads();
        cur = nxt;
    }
}

// ---------------------------------------------------------------------------
// Launch. Inputs: x0, us, A, B1, B2, C1, D11, D12, C2, D21 (all f32).
// ---------------------------------------------------------------------------
extern "C" void kernel_launch(void* const* d_in, const int* in_sizes, int n_in,
                              void* d_out, int out_size, void* d_ws, size_t ws_size,
                              hipStream_t stream) {
    (void)in_sizes; (void)n_in; (void)out_size;
    const float* x0  = (const float*)d_in[0];
    const float* us  = (const float*)d_in[1];
    const float* A   = (const float*)d_in[2];
    const float* B1  = (const float*)d_in[3];
    const float* B2  = (const float*)d_in[4];
    const float* C1  = (const float*)d_in[5];
    const float* D11 = (const float*)d_in[6];
    const float* D12 = (const float*)d_in[7];
    const float* C2  = (const float*)d_in[8];
    const float* D21 = (const float*)d_in[9];

    __bf16* ws = (__bf16*)d_ws;
    size_t off = 0;
    auto conv = [&](const float* src, size_t elems) -> __bf16* {
        __bf16* dst = ws + off;
        int nn = (int)elems;
        f32_to_bf16_kernel<<<(nn + 255) / 256, 256, 0, stream>>>(src, dst, nn);
        off += elems;
        return dst;
    };
    __bf16* Aw   = conv(A,   (size_t)NX * NX);
    __bf16* B1w  = conv(B1,  (size_t)NX * NU);
    __bf16* B2w  = conv(B2,  (size_t)NX * NW);
    __bf16* C1w  = conv(C1,  (size_t)NY * NX);
    __bf16* D11w = conv(D11, (size_t)NY * NU);
    __bf16* D12w = conv(D12, (size_t)NY * NW);
    __bf16* C2w  = conv(C2,  (size_t)NW * NX);
    __bf16* D21w = conv(D21, (size_t)NW * NU);

    const size_t xElems = (size_t)N_T * B_SZ * NX;  // x_k spill, bf16
    const size_t wElems = (size_t)N_T * B_SZ * NW;  // w_k spill, bf16
    const size_t need   = (off + xElems + wElems) * sizeof(__bf16);

    if (ws_size >= need) {
        __bf16* Xst = ws + off;
        __bf16* Wst = Xst + xElems;
        lure_rec_kernel<<<B_SZ / BM, NTHREADS, 0, stream>>>(
            x0, us, Aw, B1w, B2w, C2w, D21w, Xst, Wst);
        dim3 grid(B_SZ / BM, N_T);
        lure_y_kernel<<<grid, NTHREADS, 0, stream>>>(
            us, Xst, Wst, C1w, D11w, D12w, (float*)d_out);
    } else {
        lure_fused_kernel<<<B_SZ / BM, NTHREADS, 0, stream>>>(
            x0, us, (float*)d_out, Aw, B1w, B2w, C1w, D11w, D12w, C2w, D21w);
    }
}
